// Lion17Decoder_22273700397753
// MI455X (gfx1250) — matrix operand order repaired and checked by reading, not compiled, not run on hardware
//
#include <hip/hip_runtime.h>
#include <stdint.h>

typedef unsigned short u16;
typedef unsigned int   u32;

typedef __attribute__((ext_vector_type(16))) __bf16 v16bf;
typedef __attribute__((ext_vector_type(8)))  float  v8f;
typedef __attribute__((ext_vector_type(4)))  int    v4i_t;

// Problem constants (from reference): D=256, B=64, S=128, 4D=1024
constexpr int kD  = 256;
constexpr int kB  = 64;
constexpr int kS  = 128;
constexpr int kN4 = 1024;   // 4*D
constexpr int kK2 = 512;    // 2*D  ([dec_h | h] concatenated K)
constexpr int BPW = 16;     // batches per workgroup (matches WMMA M=16)
constexpr int AGLD = 520;   // padded LDS row stride (u16); 1040B row = 65*16B

#if defined(__gfx1250__) && __has_builtin(__builtin_amdgcn_global_load_async_to_lds_b128)
#define HAVE_ASYNC_LDS 1
#else
#define HAVE_ASYNC_LDS 0
#endif

// float -> bf16 bits, round-to-nearest-even (used only in one-time conversions
// and the small per-step h_new write)
__device__ __forceinline__ u16 f2bf_bits(float f) {
  union { float f; u32 u; } c; c.f = f;
  u32 r = c.u + 0x7FFFu + ((c.u >> 16) & 1u);
  return (u16)(r >> 16);
}

__device__ __forceinline__ float sigm(float x) { return 1.0f / (1.0f + __expf(-x)); }

// Native CDNA5 transcendental tanh (V_TANH_F32). v_nop covers the TRANS RAW
// hazard (ISA 7.4: 1 independent op after a trans op before output use).
__device__ __forceinline__ float fast_tanh(float x) {
  float y;
  asm volatile("v_tanh_f32 %0, %1\n\tv_nop" : "=v"(y) : "v"(x));
  return y;
}

// ---- WMMA fragment builders (CDNA5 16x16x32 bf16 layouts, ISA 7.12.2) ----
// A (16x32, bf16): lane L -> m=L%16, khalf=L/16; VGPR r: k=(r<4?0:16)+khalf*8+2*(r%4)+pair
__device__ __forceinline__ v16bf load_a_frag_u16(const u16* src, int ld, int lane) {
  int m = lane & 15, kh = lane >> 4;
  union { v16bf v; u32 u[8]; } A;
  const u16* row = src + m * ld;
#pragma unroll
  for (int r = 0; r < 8; ++r) {
    int k = ((r & 4) ? 16 : 0) + kh * 8 + (r & 3) * 2;
    A.u[r] = *(const u32*)(row + k);
  }
  return A.v;
}

// B (32x16, bf16) built from row-major W so B[k][n] = W[n][k]:
// lane L -> n=L%16, khalf=L/16; VGPR r: k=(r<4?0:16)+khalf*8+2*(r%4)+pair
__device__ __forceinline__ v16bf load_bT_frag_u16(const u16* W, int ld, int lane) {
  int n = lane & 15, kh = lane >> 4;
  union { v16bf v; u32 u[8]; } Bf;
  const u16* row = W + n * ld;
#pragma unroll
  for (int r = 0; r < 8; ++r) {
    int k = ((r & 4) ? 16 : 0) + kh * 8 + (r & 3) * 2;
    Bf.u[r] = *(const u32*)(row + k);
  }
  return Bf.v;
}

#define WMMA_BF16(a, b, c) \
  __builtin_amdgcn_wmma_f32_16x16x32_bf16(false, (a), false, (b), (short)0, (c), false, false)

// =====================================================================
// Kernel 1: one-time bf16 conversions into workspace.
// Wg[n][k]  (1024x512): k<256 -> W_ih[n][k], else W_hh[n][k-256]
// Wq[h][d]  (256x256) : W1[h][d]        (query projection)
// Web[h][d] (256x256) : W1[h][256+d]    (encoder projection)
// encb      (B*S*D)   : bf16 copy of encoder_outputs
// =====================================================================
__global__ void convert_inputs_kernel(const float* __restrict__ W_ih,
                                      const float* __restrict__ W_hh,
                                      const float* __restrict__ W1,
                                      const float* __restrict__ enc,
                                      u16* __restrict__ Wg, u16* __restrict__ Wq,
                                      u16* __restrict__ Web, u16* __restrict__ encb) {
  int i = blockIdx.x * 256 + threadIdx.x;
  if (i < kN4 * kK2) {
    int n = i >> 9, k = i & 511;
    float v = (k < kD) ? W_ih[n * kD + k] : W_hh[n * kD + (k - kD)];
    Wg[i] = f2bf_bits(v);
  }
  if (i < kD * kD) {
    int h = i >> 8, d = i & 255;
    Wq[i]  = f2bf_bits(W1[h * (2 * kD) + d]);
    Web[i] = f2bf_bits(W1[h * (2 * kD) + kD + d]);
  }
  if (i < kB * kS * kD) encb[i] = f2bf_bits(enc[i]);
}

// =====================================================================
// Kernel 2: enc_proj[b*S+s][h] = enc[b][s][:] . We[h][:] + b1[h]
// M = B*S = 8192, N = 256, K = 256. One 16x16 WMMA tile per wave.
// All operands pre-converted bf16 -> pure b128 loads + WMMA, no cvt VALU.
// =====================================================================
__global__ __launch_bounds__(256) void encproj_kernel(const u16* __restrict__ encb,
                                                      const u16* __restrict__ Web,
                                                      const float* __restrict__ b1,
                                                      float* __restrict__ enc_proj) {
  int wave = threadIdx.x >> 5, lane = threadIdx.x & 31;
  int tile = blockIdx.x * 8 + wave;          // 8192 tiles total
  int mt = tile >> 4, nt = tile & 15;
  const u16* Abase = encb + (size_t)mt * 16 * kD;
  const u16* Bbase = Web + (size_t)nt * 16 * kD;
  v8f acc = {};
#pragma unroll
  for (int kt = 0; kt < kD / 32; ++kt) {
    v16bf a = load_a_frag_u16(Abase + kt * 32, kD, lane);
    v16bf b = load_bT_frag_u16(Bbase + kt * 32, kD, lane);
    acc = WMMA_BF16(a, b, acc);
  }
  int n = nt * 16 + (lane & 15);
  float bias = b1[n];
#pragma unroll
  for (int v = 0; v < 8; ++v) {
    int m = mt * 16 + (lane >> 4) * 8 + v;
    enc_proj[(size_t)m * kD + n] = acc[v] + bias;
  }
}

// =====================================================================
// Kernel 3: persistent decoder. 4 workgroups x 256 threads (8 wave32).
// Each WG owns 16 batches and runs all 128 sequential steps.
// =====================================================================
__global__ __launch_bounds__(256, 1)
void decode_kernel(const float* __restrict__ enc, const u16* __restrict__ encb,
                   const float* __restrict__ b_ih, const float* __restrict__ b_hh,
                   const float* __restrict__ W2,   const float* __restrict__ b2,
                   const u16* __restrict__ Wg, const u16* __restrict__ Wq,
                   const float* __restrict__ enc_proj,
                   int* __restrict__ out_chosen, float* __restrict__ out_lp) {
  __shared__ __align__(16) u16  Ag[BPW][AGLD];   // [dec_h | h] bf16, padded stride
  __shared__ float cs[BPW][kD];                  // LSTM cell state (f32)
  __shared__ float qv[BPW][kD];                  // query projection result
  __shared__ float sc[BPW][kS];                  // attention scores
  __shared__ float w2s[kD];
  __shared__ u32   msched[BPW][4];               // scheduled bitmask (128 bits/batch)
  __shared__ u32   mmask[BPW][4];                // precedence bitmask
  __shared__ int   chosen_s[BPW];
  __shared__ float b2s;

  const int t = threadIdx.x;
  const int wave = t >> 5, lane = t & 31;
  const int b0 = blockIdx.x * BPW;

  // ---- init: dec_h0 = mean_s enc[b][s][:], h0 = 0, c0 = 0 ----
  for (int i = t; i < BPW * kD; i += 256) {
    int m = i / kD, d = i % kD;
    const float* p = enc + ((size_t)(b0 + m) * kS) * kD + d;
    float acc = 0.f;
    for (int s = 0; s < kS; ++s) acc += p[s * kD];
    Ag[m][d]      = f2bf_bits(acc * (1.0f / kS));
    Ag[m][kD + d] = 0;            // h0 = 0 (bf16 zero bits)
    cs[m][d]      = 0.f;
  }
  for (int i = t; i < kD; i += 256) w2s[i] = W2[i];
  if (t < BPW * 4) { msched[t >> 2][t & 3] = 0u; mmask[t >> 2][t & 3] = 0xFEFEFEFEu; } // ops>0
  if (t == 0) b2s = b2[0];

  // Hoist per-thread LSTM biases: wave owns d in [32*wave, 32*wave+32)
  float biasv[2][4];
  {
    int nn = lane & 15;
#pragma unroll
    for (int j = 0; j < 2; ++j) {
      int d = (2 * wave + j) * 16 + nn;
#pragma unroll
      for (int g = 0; g < 4; ++g) biasv[j][g] = b_ih[g * kD + d] + b_hh[g * kD + d];
    }
  }
  __syncthreads();

  for (int step = 0; step < kS; ++step) {
    // ---------- Phase 1: gates = [dec_h|h] @ Wg^T  (M=16,N=1024,K=512) ----------
    // Wave w computes n-tiles {g*16 + 2w + j} for g=0..3, j=0..1 -> i/f/g/o for
    // d in [32w, 32w+32): LSTM nonlinearity stays in registers.
    v8f acc[2][4];
#pragma unroll
    for (int j = 0; j < 2; ++j)
#pragma unroll
      for (int g = 0; g < 4; ++g) acc[j][g] = (v8f){};

    for (int kt = 0; kt < kK2 / 32; ++kt) {
      v16bf a = load_a_frag_u16(&Ag[0][0] + kt * 32, AGLD, lane);
      if (kt + 1 < kK2 / 32)   // keep next weight slab warm (global_prefetch_b8)
        __builtin_prefetch(Wg + (size_t)(2 * wave) * 16 * kK2 + (kt + 1) * 32, 0, 0);
#pragma unroll
      for (int j = 0; j < 2; ++j) {
#pragma unroll
        for (int g = 0; g < 4; ++g) {
          int nt = g * 16 + 2 * wave + j;
          v16bf b = load_bT_frag_u16(Wg + (size_t)nt * 16 * kK2 + kt * 32, kK2, lane);
          acc[j][g] = WMMA_BF16(a, b, acc[j][g]);
        }
      }
    }
    __syncthreads();  // all waves done reading Ag before h-half is overwritten

    // ---------- Phase 2: LSTM pointwise update (registers -> cs, Ag h-half) ----
#pragma unroll
    for (int j = 0; j < 2; ++j) {
      int d = (2 * wave + j) * 16 + (lane & 15);
#pragma unroll
      for (int v = 0; v < 8; ++v) {
        int m = (lane >> 4) * 8 + v;
        float iv = acc[j][0][v] + biasv[j][0];
        float fv = acc[j][1][v] + biasv[j][1];
        float gv = acc[j][2][v] + biasv[j][2];
        float ov = acc[j][3][v] + biasv[j][3];
        float cnew = sigm(fv) * cs[m][d] + sigm(iv) * fast_tanh(gv);
        float hnew = sigm(ov) * fast_tanh(cnew);
        cs[m][d] = cnew;
        Ag[m][kD + d] = f2bf_bits(hnew);   // feeds both next gates GEMM and q GEMM
      }
    }
    __syncthreads();

    // ---------- Phase 3: q = h_new @ Wq^T  (M=16,N=256,K=256) ----------
    v8f qa[2]; qa[0] = (v8f){}; qa[1] = (v8f){};
    for (int kt = 0; kt < kD / 32; ++kt) {
      v16bf a = load_a_frag_u16(&Ag[0][kD] + kt * 32, AGLD, lane);
#pragma unroll
      for (int j = 0; j < 2; ++j) {
        int nt = 2 * wave + j;
        v16bf b = load_bT_frag_u16(Wq + (size_t)nt * 16 * kD + kt * 32, kD, lane);
        qa[j] = WMMA_BF16(a, b, qa[j]);
      }
    }
#pragma unroll
    for (int j = 0; j < 2; ++j) {
      int n = (2 * wave + j) * 16 + (lane & 15);
#pragma unroll
      for (int v = 0; v < 8; ++v) qv[(lane >> 4) * 8 + v][n] = qa[j][v];
    }
    __syncthreads();

    // ---------- Phase 4: attention scores + masking ----------
    // 16 batches x 128 slots = 2048 dot products of length 256, native v_tanh.
#pragma unroll
    for (int p = 0; p < 8; ++p) {
      int idx = p * 256 + t;
      int m = idx >> 7, s = idx & 127;
      const float4* ep = (const float4*)(enc_proj + (((size_t)(b0 + m) * kS) + s) * kD);
      const float4* qp = (const float4*)(&qv[m][0]);
      const float4* wp = (const float4*)(&w2s[0]);
      float dot = 0.f;
      for (int dd = 0; dd < kD / 4; ++dd) {
        float4 e = ep[dd], q = qp[dd], w = wp[dd];
        dot += fast_tanh(e.x + q.x) * w.x + fast_tanh(e.y + q.y) * w.y +
               fast_tanh(e.z + q.z) * w.z + fast_tanh(e.w + q.w) * w.w;
      }
      u32 dead = ((msched[m][s >> 5] | mmask[m][s >> 5]) >> (s & 31)) & 1u;
      sc[m][s] = dead ? -__builtin_inff() : (dot + b2s);
    }
    __syncthreads();

    // ---------- Phase 5: per-batch argmax + log-softmax + mask update ----------
#pragma unroll
    for (int mi = 0; mi < 2; ++mi) {
      int mm = 2 * wave + mi;
      float vals[4];
      float bestv = -__builtin_inff(); int besti = kS;
#pragma unroll
      for (int r = 0; r < 4; ++r) {
        int s = lane + 32 * r;
        float v = sc[mm][s]; vals[r] = v;
        if (v > bestv || (v == bestv && s < besti)) { bestv = v; besti = s; }
      }
      for (int off = 16; off; off >>= 1) {
        float ov = __shfl_xor(bestv, off, 32);
        int   oi = __shfl_xor(besti, off, 32);
        if (ov > bestv || (ov == bestv && oi < besti)) { bestv = ov; besti = oi; }
      }
      float sum = 0.f;
#pragma unroll
      for (int r = 0; r < 4; ++r) sum += __expf(vals[r] - bestv);
      for (int off = 16; off; off >>= 1) sum += __shfl_xor(sum, off, 32);
      if (lane == 0) {
        float lse = bestv + __logf(sum);
        int b = b0 + mm;
        out_chosen[(size_t)b * kS + step] = besti;       // chosen_seq.T
        out_lp[(size_t)b * kS + step]     = bestv - lse; // logp_seq.T
        chosen_s[mm] = besti;
        msched[mm][besti >> 5] |= 1u << (besti & 31);
        int co = besti & 7;                              // op = s % 8 (fixed structure)
        if (co < 7) { int u = besti + 1; mmask[mm][u >> 5] &= ~(1u << (u & 31)); }
      }
    }
    __syncthreads();

    // ---------- Phase 6: dec_h_new = encb[b][chosen] -> Ag dec-half ----------
    // bf16 source: pure copy. Async-to-LDS path when the toolchain exposes it:
    // one GLOBAL_LOAD_ASYNC_TO_LDS_B128 per wave per row (32 lanes x 16B = 512B).
#if HAVE_ASYNC_LDS
#pragma unroll
    for (int mi = 0; mi < 2; ++mi) {
      int m = 2 * wave + mi;
      const u16* g = encb + (((size_t)(b0 + m) * kS) + chosen_s[m]) * kD + lane * 8;
      u16* l = &Ag[m][0] + lane * 8;
      __builtin_amdgcn_global_load_async_to_lds_b128(
          (v4i_t*)(uintptr_t)g,
          (__attribute__((address_space(3))) v4i_t*)(uintptr_t)l,
          0, 0);
    }
    asm volatile("s_wait_asynccnt 0" ::: "memory");
#else
#pragma unroll
    for (int mi = 0; mi < 2; ++mi) {
      int m = 2 * wave + mi;
      const u32* g = (const u32*)(encb + (((size_t)(b0 + m) * kS) + chosen_s[m]) * kD);
      u32* l = (u32*)&Ag[m][0];
      for (int r = lane; r < kD / 2; r += 32) l[r] = g[r];
    }
#endif
    __syncthreads();
  }
}

// =====================================================================
// Host launcher. Inputs (setup_inputs order):
// 0 encoder_outputs (B,S,D) f32   1 S_seq (B,S,2) i32 (structure is fixed)
// 2 W_ih (4D,D)  3 W_hh (4D,D)  4 b_ih (4D)  5 b_hh (4D)
// 6 W1 (D,2D)    7 b1 (D)       8 W2 (1,D)   9 b2 (1)
// Output: chosen_seq.T (B,S) int32 bits, then logp_seq.T (B,S) f32,
// concatenated flat in return order (out_size = 2*B*S = 16384).
// Workspace: [0,8M) enc_proj f32 ; [8M,9M) Wg bf16 ; [9M,+128K) Wq bf16 ;
//            [9.25M,+128K) We bf16 ; [9.5M,+4M) enc bf16.  Total 13.5 MB.
// =====================================================================
extern "C" void kernel_launch(void* const* d_in, const int* in_sizes, int n_in,
                              void* d_out, int out_size, void* d_ws, size_t ws_size,
                              hipStream_t stream) {
  const float* enc  = (const float*)d_in[0];
  const float* W_ih = (const float*)d_in[2];
  const float* W_hh = (const float*)d_in[3];
  const float* b_ih = (const float*)d_in[4];
  const float* b_hh = (const float*)d_in[5];
  const float* W1   = (const float*)d_in[6];
  const float* b1   = (const float*)d_in[7];
  const float* W2   = (const float*)d_in[8];
  const float* b2   = (const float*)d_in[9];

  char* ws = (char*)d_ws;
  float* enc_proj = (float*)ws;                                     // 8 MB
  u16*   Wg   = (u16*)(ws + (size_t)8 * 1024 * 1024);               // 1 MB
  u16*   Wq   = (u16*)(ws + (size_t)9 * 1024 * 1024);               // 128 KB
  u16*   Web  = (u16*)(ws + (size_t)9 * 1024 * 1024 + 256 * 1024);  // 128 KB
  u16*   encb = (u16*)(ws + (size_t)9 * 1024 * 1024 + 512 * 1024);  // 4 MB

  int*   out_chosen = (int*)d_out;
  float* out_lp     = (float*)d_out + (size_t)kB * kS;

  convert_inputs_kernel<<<(kB * kS * kD + 255) / 256, 256, 0, stream>>>(
      W_ih, W_hh, W1, enc, Wg, Wq, Web, encb);
  encproj_kernel<<<(kB * kS / 16) * 16 / 8, 256, 0, stream>>>(encb, Web, b1, enc_proj);
  decode_kernel<<<kB / BPW, 256, 0, stream>>>(enc, encb, b_ih, b_hh, W2, b2, Wg, Wq,
                                              enc_proj, out_chosen, out_lp);
}
